// LinearAttentionTransformer_55911884259786
// MI455X (gfx1250) — matrix operand order repaired and checked by reading, not compiled, not run on hardware
//
#include <hip/hip_runtime.h>

#define LN_EPS 1e-5f
#define LEAKY_SLOPE 0.01f

typedef __attribute__((ext_vector_type(16))) __bf16 v16bf;
typedef __attribute__((ext_vector_type(8)))  __bf16 v8bf;
typedef __attribute__((ext_vector_type(8)))  float  v8f;

union V16U { v16bf v; v8bf h[2]; uint4 u[2]; };

enum { EPI_BF16 = 0, EPI_BIAS_RES = 1, EPI_RES = 2, EPI_LEAKY = 3 };

// ---------------------------------------------------------------------------
// LayerNorm: one block per row of 1024 fp32, writes bf16.
// ---------------------------------------------------------------------------
__global__ __launch_bounds__(256) void ln_kernel(
    const float* __restrict__ x, const float* __restrict__ w,
    const float* __restrict__ b, __bf16* __restrict__ o)
{
    __shared__ float red[256];
    const int tid = threadIdx.x;
    const size_t base = (size_t)blockIdx.x * 1024 + (size_t)tid * 4;
    float4 v = *(const float4*)(x + base);
    red[tid] = v.x + v.y + v.z + v.w;
    __syncthreads();
    for (int st = 128; st; st >>= 1) { if (tid < st) red[tid] += red[tid + st]; __syncthreads(); }
    float mean = red[0] * (1.0f / 1024.0f);
    __syncthreads();
    float dx = v.x - mean, dy = v.y - mean, dz = v.z - mean, dw = v.w - mean;
    red[tid] = dx * dx + dy * dy + dz * dz + dw * dw;
    __syncthreads();
    for (int st = 128; st; st >>= 1) { if (tid < st) red[tid] += red[tid + st]; __syncthreads(); }
    float r = rsqrtf(red[0] * (1.0f / 1024.0f) + LN_EPS);
    float4 wv = *(const float4*)(w + tid * 4);
    float4 bv = *(const float4*)(b + tid * 4);
    o[base + 0] = (__bf16)(dx * r * wv.x + bv.x);
    o[base + 1] = (__bf16)(dy * r * wv.y + bv.y);
    o[base + 2] = (__bf16)(dz * r * wv.z + bv.z);
    o[base + 3] = (__bf16)(dw * r * wv.w + bv.w);
}

// ---------------------------------------------------------------------------
// Softmax over feature dim (128 per head), one warp per (row, head); in-place.
// ---------------------------------------------------------------------------
__global__ __launch_bounds__(256) void softmax_q_kernel(__bf16* __restrict__ q)
{
    const int warp = threadIdx.x >> 5, lane = threadIdx.x & 31;
    const long long seg = (long long)blockIdx.x * 8 + warp;  // row*8 + head
    const long long row = seg >> 3;
    const int h = (int)(seg & 7);
    __bf16* p = q + row * 1024 + h * 128 + lane * 4;
    float f[4];
#pragma unroll
    for (int j = 0; j < 4; j++) f[j] = (float)p[j];
    float mx = fmaxf(fmaxf(f[0], f[1]), fmaxf(f[2], f[3]));
    for (int o = 16; o; o >>= 1) mx = fmaxf(mx, __shfl_xor(mx, o, 32));
    float s = 0.f;
#pragma unroll
    for (int j = 0; j < 4; j++) { f[j] = __expf(f[j] - mx); s += f[j]; }
    for (int o = 16; o; o >>= 1) s += __shfl_xor(s, o, 32);
    float inv = 1.0f / s;
#pragma unroll
    for (int j = 0; j < 4; j++) p[j] = (__bf16)(f[j] * inv);
}

// ---------------------------------------------------------------------------
// Softmax over sequence dim (4096), one block per (batch, channel); in-place.
// ---------------------------------------------------------------------------
__global__ __launch_bounds__(256) void softmax_k_kernel(__bf16* __restrict__ k)
{
    __shared__ float red[256];
    const int tid = threadIdx.x;
    const int b = blockIdx.x >> 10;
    const int c = blockIdx.x & 1023;
    const size_t base = (size_t)b * 4096 * 1024 + c;
    float f[16];
#pragma unroll
    for (int j = 0; j < 16; j++)
        f[j] = (float)k[base + (size_t)(tid + j * 256) * 1024];
    float mx = -3.0e38f;
#pragma unroll
    for (int j = 0; j < 16; j++) mx = fmaxf(mx, f[j]);
    red[tid] = mx; __syncthreads();
    for (int st = 128; st; st >>= 1) { if (tid < st) red[tid] = fmaxf(red[tid], red[tid + st]); __syncthreads(); }
    mx = red[0]; __syncthreads();
    float s = 0.f;
#pragma unroll
    for (int j = 0; j < 16; j++) { f[j] = __expf(f[j] - mx); s += f[j]; }
    red[tid] = s; __syncthreads();
    for (int st = 128; st; st >>= 1) { if (tid < st) red[tid] += red[tid + st]; __syncthreads(); }
    float inv = 1.0f / red[0];
#pragma unroll
    for (int j = 0; j < 16; j++)
        k[base + (size_t)(tid + j * 256) * 1024] = (__bf16)(f[j] * inv);
}

// ---------------------------------------------------------------------------
// Tiled WMMA GEMM: C[M,N] = A[M,K] (bf16) * B[K,N] (TB -> bf16), fp32 accum.
// Block tile 128x128, K-step 32, 8 waves each owning 32x64 (2x4 WMMA tiles).
// A tile staged with GLOBAL_LOAD_ASYNC_TO_LDS_B128 (ASYNCcnt, inline asm);
// B tile staged via VGPRs with fp32->bf16 convert, k-index spread across
// lanes so concurrent ds_store_b16s hit distinct banks.
// TRANSA: A is stored [K][M] in global (used for K^T V).
// Batched via blockIdx.z = b*Hn + h with independent element strides.
// ---------------------------------------------------------------------------
template <typename TB, bool TRANSA, int EPI>
__global__ __launch_bounds__(256) void gemm_kernel(
    const __bf16* __restrict__ A, const TB* __restrict__ Bm,
    float* __restrict__ Cf, __bf16* __restrict__ Cbf,
    const float* __restrict__ bias, const float* __restrict__ Res,
    int M, int N, int K, int lda, int ldb, int ldc,
    long long sAb, long long sAh, long long sBb, long long sBh,
    long long sCb, long long sCh, int Hn)
{
    constexpr int BM = 128, BN = 128, BK = 32;
    constexpr int LSA = 40, LSB = 40;  // padded strides (80B)
    __shared__ __attribute__((aligned(16))) __bf16 As[BM * LSA];
    __shared__ __attribute__((aligned(16))) __bf16 Bs[BN * LSB];

    const int tid = threadIdx.x;
    const int zb = blockIdx.z / Hn, zh = blockIdx.z % Hn;
    const long long offA = (long long)zb * sAb + (long long)zh * sAh;
    const long long offB = (long long)zb * sBb + (long long)zh * sBh;
    const long long offC = (long long)zb * sCb + (long long)zh * sCh;
    const int m0 = blockIdx.y * BM, n0 = blockIdx.x * BN;

    const int wave = tid >> 5, lane = tid & 31;
    const int wm = (wave >> 1) * 32, wn = (wave & 1) * 64;
    const int lh = lane >> 4;       // half-wave select
    const int lm = lane & 15;       // row/col within 16

    // Staging indices: k spread across lanes -> conflict-free(ish) ds stores.
    const int skk = tid & 31;            // k within tile (0..31)
    const int sns = (tid >> 5) * 16;     // col/row segment (0..112)

    // Async A staging indices (non-trans): 2 threads per row of 32 halves.
    const int ar = tid >> 1, asg = (tid & 1) * 16;
    const unsigned lds_a = (unsigned)(unsigned long long)(void*)&As[ar * LSA + asg];

    v8f acc[2][4];
#pragma unroll
    for (int mi = 0; mi < 2; mi++)
#pragma unroll
        for (int ni = 0; ni < 4; ni++)
#pragma unroll
            for (int r = 0; r < 8; r++) acc[mi][ni][r] = 0.0f;

    for (int k0 = 0; k0 < K; k0 += BK) {
        // ---- stage A tile (128 x 32 bf16) ----
        if constexpr (!TRANSA) {
            const __bf16* gp = A + offA + (long long)(m0 + ar) * lda + k0 + asg;
            // INST_OFFSET applies to both LDS and global addresses.
            asm volatile(
                "global_load_async_to_lds_b128 %0, %1, off\n\t"
                "global_load_async_to_lds_b128 %0, %1, off offset:16"
                :: "v"(lds_a), "v"(gp) : "memory");
        } else {
            const __bf16* gp = A + offA + (long long)(k0 + skk) * lda + m0 + sns;
            V16U t;
            t.u[0] = *(const uint4*)gp;
            t.u[1] = *(const uint4*)(gp + 8);
#pragma unroll
            for (int j = 0; j < 16; j++) As[(sns + j) * LSA + skk] = t.v[j];
        }
        // ---- stage B tile (32 x 128 -> column-major bf16 [n][k]) ----
        {
            const TB* gp = Bm + offB + (long long)(k0 + skk) * ldb + n0 + sns;
            if constexpr (sizeof(TB) == 4) {
                float4 f0 = *(const float4*)(gp + 0);
                float4 f1 = *(const float4*)(gp + 4);
                float4 f2 = *(const float4*)(gp + 8);
                float4 f3 = *(const float4*)(gp + 12);
                if (k0 + BK < K)  // prefetch next k-tile of weights into L2
                    __builtin_prefetch((const void*)(gp + (long long)BK * ldb), 0, 3);
                float fv[16] = { f0.x, f0.y, f0.z, f0.w, f1.x, f1.y, f1.z, f1.w,
                                 f2.x, f2.y, f2.z, f2.w, f3.x, f3.y, f3.z, f3.w };
#pragma unroll
                for (int j = 0; j < 16; j++) Bs[(sns + j) * LSB + skk] = (__bf16)fv[j];
            } else {
                V16U t;
                t.u[0] = *(const uint4*)gp;
                t.u[1] = *(const uint4*)((const __bf16*)gp + 8);
#pragma unroll
                for (int j = 0; j < 16; j++) Bs[(sns + j) * LSB + skk] = t.v[j];
            }
        }
        if constexpr (!TRANSA) {
            asm volatile("s_wait_asynccnt 0x0" ::: "memory");
        }
        __syncthreads();

        // ---- fragments: contiguous k = 16*lh + s, same for A and B ----
        V16U a[2], bf[4];
#pragma unroll
        for (int mi = 0; mi < 2; mi++) {
            const __bf16* ap = &As[(wm + mi * 16 + lm) * LSA + lh * 16];
            a[mi].h[0] = *(const v8bf*)ap;          // K = 16*lh..16*lh+7
            a[mi].h[1] = *(const v8bf*)(ap + 8);    // K = 16*lh+8..16*lh+15
        }
#pragma unroll
        for (int ni = 0; ni < 4; ni++) {
            const __bf16* bp = &Bs[(wn + ni * 16 + lm) * LSB + lh * 16];
            bf[ni].h[0] = *(const v8bf*)bp;         // K run of 16 per half-wave
            bf[ni].h[1] = *(const v8bf*)(bp + 8);
        }
#pragma unroll
        for (int mi = 0; mi < 2; mi++)
#pragma unroll
            for (int ni = 0; ni < 4; ni++)
                acc[mi][ni] = __builtin_amdgcn_wmma_f32_16x16x32_bf16(
                    false, a[mi].v, false, bf[ni].v, (short)0, acc[mi][ni],
                    false, false);
        __syncthreads();
    }

    // ---- epilogue: C/D layout = VGPR r, lanes 0-15 M=r N=lane, lanes 16-31 M=r+8 ----
    const int nl = lane & 15;
    const int mb = (lane >> 4) * 8;
#pragma unroll
    for (int mi = 0; mi < 2; mi++)
#pragma unroll
        for (int ni = 0; ni < 4; ni++) {
            const int gm0 = m0 + wm + mi * 16 + mb;
            const int gn = n0 + wn + ni * 16 + nl;
#pragma unroll
            for (int r = 0; r < 8; r++) {
                const long long idx = offC + (long long)(gm0 + r) * ldc + gn;
                float val = acc[mi][ni][r];
                if constexpr (EPI == EPI_BF16) {
                    Cbf[idx] = (__bf16)val;
                } else if constexpr (EPI == EPI_BIAS_RES) {
                    Cf[idx] = val + bias[gn] + Res[idx];
                } else if constexpr (EPI == EPI_RES) {
                    Cf[idx] = val + Res[idx];
                } else {  // EPI_LEAKY
                    val = (val >= 0.0f) ? val : LEAKY_SLOPE * val;
                    Cbf[idx] = (__bf16)val;
                }
            }
        }
}

// ---------------------------------------------------------------------------
extern "C" void kernel_launch(void* const* d_in, const int* in_sizes, int n_in,
                              void* d_out, int out_size, void* d_ws, size_t ws_size,
                              hipStream_t stream)
{
    const float* x    = (const float*)d_in[0];
    const float* ln1w = (const float*)d_in[1];
    const float* ln1b = (const float*)d_in[2];
    const float* wq   = (const float*)d_in[3];
    const float* wk   = (const float*)d_in[4];
    const float* wv   = (const float*)d_in[5];
    const float* wo   = (const float*)d_in[6];
    const float* bo   = (const float*)d_in[7];
    const float* ln2w = (const float*)d_in[8];
    const float* ln2b = (const float*)d_in[9];
    const float* w1   = (const float*)d_in[10];
    const float* w2   = (const float*)d_in[11];

    const int Bb = 2, T = 4096, E = 1024, H = 8, DH = 128, FF = 4096;
    const int BT = Bb * T;  // 8192

    char* ws = (char*)d_ws;
    float*  x_buf = (float*)ws;  ws += (size_t)BT * E * 4;
    __bf16* xn    = (__bf16*)ws; ws += (size_t)BT * E * 2;
    __bf16* qb    = (__bf16*)ws; ws += (size_t)BT * E * 2;
    __bf16* kb    = (__bf16*)ws; ws += (size_t)BT * E * 2;
    __bf16* vb    = (__bf16*)ws; ws += (size_t)BT * E * 2;
    __bf16* ctx   = (__bf16*)ws; ws += (size_t)Bb * H * DH * DH * 2;
    __bf16* att   = (__bf16*)ws; ws += (size_t)BT * E * 2;
    __bf16* hdn   = (__bf16*)ws; ws += (size_t)BT * FF * 2;

    const float* xsrc = x;  // residual / LN source (fp32)
    for (int i = 0; i < 6; i++) {
        // PreNorm 1
        ln_kernel<<<BT, 256, 0, stream>>>(xsrc, ln1w + (size_t)i * E,
                                          ln1b + (size_t)i * E, xn);
        // Q, K, V projections  (8192 x 1024 x 1024)
        gemm_kernel<float, false, EPI_BF16><<<dim3(E / 128, BT / 128, 1), 256, 0, stream>>>(
            xn, wq + (size_t)i * E * E, nullptr, qb, nullptr, nullptr,
            BT, E, E, E, E, E, 0, 0, 0, 0, 0, 0, 1);
        gemm_kernel<float, false, EPI_BF16><<<dim3(E / 128, BT / 128, 1), 256, 0, stream>>>(
            xn, wk + (size_t)i * E * E, nullptr, kb, nullptr, nullptr,
            BT, E, E, E, E, E, 0, 0, 0, 0, 0, 0, 1);
        gemm_kernel<float, false, EPI_BF16><<<dim3(E / 128, BT / 128, 1), 256, 0, stream>>>(
            xn, wv + (size_t)i * E * E, nullptr, vb, nullptr, nullptr,
            BT, E, E, E, E, E, 0, 0, 0, 0, 0, 0, 1);
        // softmax(q) over feature dim, softmax(k) over sequence dim (in-place)
        softmax_q_kernel<<<BT, 256, 0, stream>>>(qb);
        softmax_k_kernel<<<Bb * E, 256, 0, stream>>>(kb);
        // context[b,h] = k^T v : 128 x 128 x 4096 per (b,h)
        gemm_kernel<__bf16, true, EPI_BF16><<<dim3(1, 1, Bb * H), 256, 0, stream>>>(
            kb, vb, nullptr, ctx, nullptr, nullptr,
            DH, DH, T, E, E, DH,
            (long long)T * E, DH,            // A strides (b,h)
            (long long)T * E, DH,            // B strides (b,h)
            (long long)H * DH * DH, (long long)DH * DH,  // C strides
            H);
        // attn[b,h] = q @ context : 4096 x 128 x 128 per (b,h)
        gemm_kernel<__bf16, false, EPI_BF16><<<dim3(1, T / 128, Bb * H), 256, 0, stream>>>(
            qb, ctx, nullptr, att, nullptr, nullptr,
            T, DH, DH, E, DH, E,
            (long long)T * E, DH,
            (long long)H * DH * DH, (long long)DH * DH,
            (long long)T * E, DH,
            H);
        // x = attn @ wo + bo + x
        gemm_kernel<float, false, EPI_BIAS_RES><<<dim3(E / 128, BT / 128, 1), 256, 0, stream>>>(
            att, wo + (size_t)i * E * E, x_buf, nullptr,
            bo + (size_t)i * E, xsrc,
            BT, E, E, E, E, E, 0, 0, 0, 0, 0, 0, 1);
        // PreNorm 2
        ln_kernel<<<BT, 256, 0, stream>>>(x_buf, ln2w + (size_t)i * E,
                                          ln2b + (size_t)i * E, xn);
        // hdn = LeakyReLU(xn @ w1)  (8192 x 4096 x 1024)
        gemm_kernel<float, false, EPI_LEAKY><<<dim3(FF / 128, BT / 128, 1), 256, 0, stream>>>(
            xn, w1 + (size_t)i * E * FF, nullptr, hdn, nullptr, nullptr,
            BT, FF, E, E, FF, FF, 0, 0, 0, 0, 0, 0, 1);
        // x = hdn @ w2 + x  (8192 x 1024 x 4096); final layer -> d_out
        float* outp = (i == 5) ? (float*)d_out : x_buf;
        gemm_kernel<float, false, EPI_RES><<<dim3(E / 128, BT / 128, 1), 256, 0, stream>>>(
            hdn, w2 + (size_t)i * FF * E, outp, nullptr,
            nullptr, x_buf,
            BT, E, FF, FF, E, E, 0, 0, 0, 0, 0, 0, 1);
        xsrc = x_buf;
    }
    (void)in_sizes; (void)n_in; (void)out_size; (void)ws_size;
}
